// ConvLSTMPSLA_29609504538947
// MI455X (gfx1250) — matrix operand order repaired and checked by reading, not compiled, not run on hardware
//
#include <hip/hip_runtime.h>
#include <cstdint>
#include <cstddef>

// ---------------------------------------------------------------------------
// Types for CDNA5 WMMA (wave32): 16x16x32 bf16 -> f32
// ---------------------------------------------------------------------------
typedef __attribute__((ext_vector_type(16))) __bf16   v16bf;
typedef __attribute__((ext_vector_type(8)))  float    v8f;
typedef __attribute__((ext_vector_type(4)))  unsigned uint4v;

struct U8x32 { uint4v lo, hi; };   // 32 bytes, bit-castable to v16bf

__device__ __forceinline__ unsigned f2bf(float f) {
    // round-to-nearest-even fp32 -> bf16 (returned in low 16 bits)
    unsigned u = __float_as_uint(f);
    u += 0x7FFFu + ((u >> 16) & 1u);
    return u >> 16;
}

__device__ __forceinline__ float sigmoidf(float x) { return 1.0f / (1.0f + __expf(-x)); }

// ---------------------------------------------------------------------------
// Streaming f32 -> packed bf16 (4 elements / thread). Done ONCE per weight so
// the GEMM A-tiles become raw async copies and level-1 weights fit in L2.
// ---------------------------------------------------------------------------
__global__ __launch_bounds__(256) void f32_to_bf16_kernel(
    const float* __restrict__ in, unsigned short* __restrict__ out, int n4)
{
    const int i = blockIdx.x * 256 + threadIdx.x;
    if (i >= n4) return;
    const float4 v = reinterpret_cast<const float4*>(in)[i];
    uint2 pk;
    pk.x = f2bf(v.x) | (f2bf(v.y) << 16);
    pk.y = f2bf(v.z) | (f2bf(v.w) << 16);
    reinterpret_cast<uint2*>(out)[i] = pk;
}

// ---------------------------------------------------------------------------
// Attention: corr over 9 shifts -> softmax -> weighted recombination.
// One wave per pixel; lanes stride over channels; wave32 shfl reduction.
// ---------------------------------------------------------------------------
template<int C, int H>
__global__ __launch_bounds__(256) void attn_kernel(
    const float* __restrict__ feat1,   // xp  (shifted source)
    const float* __restrict__ feat2,   // x
    float* __restrict__ new_feat)
{
    constexpr int HW = H * H;
    const int lane = threadIdx.x & 31;
    const int pix  = blockIdx.x * 8 + (threadIdx.x >> 5);   // over B*HW
    const int b  = pix / HW;
    const int ph = pix - b * HW;
    const int oh = ph / H;
    const int ow = ph - oh * H;
    const float* f1 = feat1 + (size_t)b * C * HW;
    const float* f2 = feat2 + (size_t)b * C * HW;

    float corr[9];
#pragma unroll
    for (int k = 0; k < 9; ++k) corr[k] = 0.0f;

    for (int c = lane; c < C; c += 32) {
        const float v2 = f2[(size_t)c * HW + ph];
#pragma unroll
        for (int k = 0; k < 9; ++k) {
            const int ih = oh + k / 3 - 1;
            const int iw = ow + k % 3 - 1;
            const float v1 = ((unsigned)ih < (unsigned)H && (unsigned)iw < (unsigned)H)
                                 ? f1[(size_t)c * HW + ih * H + iw] : 0.0f;
            corr[k] += v2 * v1;
        }
    }
#pragma unroll
    for (int k = 0; k < 9; ++k)
#pragma unroll
        for (int off = 16; off; off >>= 1)
            corr[k] += __shfl_xor(corr[k], off, 32);

    float mx = corr[0];
#pragma unroll
    for (int k = 1; k < 9; ++k) mx = fmaxf(mx, corr[k]);
    float a[9], s = 0.0f;
#pragma unroll
    for (int k = 0; k < 9; ++k) { a[k] = __expf(corr[k] - mx); s += a[k]; }
    const float inv = 1.0f / s;
#pragma unroll
    for (int k = 0; k < 9; ++k) a[k] *= inv;

    for (int c = lane; c < C; c += 32) {
        float acc = 0.0f;
#pragma unroll
        for (int k = 0; k < 9; ++k) {
            const int ih = oh + k / 3 - 1;
            const int iw = ow + k % 3 - 1;
            const float v1 = ((unsigned)ih < (unsigned)H && (unsigned)iw < (unsigned)H)
                                 ? f1[(size_t)c * HW + ih * H + iw] : 0.0f;
            acc += a[k] * v1;
        }
        new_feat[((size_t)b * C + c) * HW + ph] = acc;
    }
}

// ---------------------------------------------------------------------------
// Implicit-GEMM via WMMA bf16:
//   out[b][m][h][w] = sum_k wA[m][k]*im2col(srcA)[k][n]
//                   + sum_k wB[m][k]*im2col(srcB)[k][n] + biasA[m] + biasB[m]
// R3 = 1: 3x3 conv (K = 9*C per source, k = c*9 + kh*3 + kw, zero pad 1)
// R3 = 0: 1x1 conv (K = C per source)
// WB16 = 1: weights pre-converted to bf16; A-tiles staged with
//           GLOBAL_LOAD_ASYNC_TO_LDS_B128 (ASYNCcnt) -- raw copy, no VALU.
// WB16 = 0: weights f32; staged synchronously with in-kernel conversion.
// Block tile: 128(M) x 128(N), BK = 64 (two WMMA K-steps per barrier round).
// 8 waves: wave tile 32(M) x 64(N) = 2x4 WMMA tiles -> 16 WMMAs / wave / round.
// ---------------------------------------------------------------------------
template<int C, int H, int R3, bool WB16>
__global__ __launch_bounds__(256) void wmma_gemm_kernel(
    const float* __restrict__ srcA, const float* __restrict__ srcB,
    const void* __restrict__ wAv,   const void* __restrict__ wBv,
    int wstride,                               // weight row stride (elements)
    const float* __restrict__ biasA, const float* __restrict__ biasB,
    int M,                                     // output channels (mult of 128)
    float* __restrict__ out)
{
    constexpr int HW = H * H;
    constexpr int KC = C * (R3 ? 9 : 1);       // K per source
    constexpr int BM = 128, BN = 128, BK = 64;

    __shared__ __align__(16) unsigned short As[BM][BK];   // [m][k] bf16 (16 KB)
    __shared__ __align__(16) unsigned short Bs[BN][BK];   // [n][k] bf16 (16 KB)

    const int tid  = threadIdx.x;
    const int lane = tid & 31;
    const int wave = tid >> 5;
    const int wm   = wave & 3;                 // 0..3  (M direction, 32 rows)
    const int wn   = wave >> 2;                // 0..1  (N direction, 64 cols)
    const int m0   = blockIdx.x * BM;
    const int n0   = blockIdx.y * BN;

    v8f acc[2][4] = {};

#pragma unroll 1
    for (int p = 0; p < 2; ++p) {
        const float* __restrict__ src = p ? srcB : srcA;
        const void*  __restrict__ wgtv = p ? wBv : wAv;
#pragma unroll 1
        for (int k0 = 0; k0 < KC; k0 += BK) {
            __syncthreads();   // previous round's readers done before restage

            // ---- Stage A (weights): 128 rows x 64 K.
            if constexpr (WB16) {
                // Raw bf16 copy: 64B/thread via 4x async global->LDS b128.
                const unsigned short* wgt = (const unsigned short*)wgtv;
                const unsigned short* wrow =
                    wgt + (size_t)(m0 + (tid >> 1)) * (size_t)wstride + k0 + (tid & 1) * 32;
                const unsigned lds_base =
                    (unsigned)(uintptr_t)&As[tid >> 1][(tid & 1) * 32];
#pragma unroll
                for (int q = 0; q < 4; ++q) {
                    const unsigned long long ga =
                        (unsigned long long)(uintptr_t)(wrow + q * 8);
                    const unsigned la = lds_base + q * 16;
                    asm volatile("global_load_async_to_lds_b128 %0, %1, off"
                                 :: "v"(la), "v"(ga) : "memory");
                }
            } else {
                const float* wgt = (const float*)wgtv;
#pragma unroll
                for (int e = tid; e < BM * BK / 4; e += 256) {
                    const int m  = e >> 4;
                    const int k4 = (e & 15) << 2;
                    const float4 w4 = *reinterpret_cast<const float4*>(
                        &wgt[(size_t)(m0 + m) * wstride + k0 + k4]);
                    uint2 pk;
                    pk.x = f2bf(w4.x) | (f2bf(w4.y) << 16);
                    pk.y = f2bf(w4.z) | (f2bf(w4.w) << 16);
                    *reinterpret_cast<uint2*>(&As[m][k4]) = pk;
                }
                if (k0 + BK < KC)
                    __builtin_prefetch(&wgt[(size_t)(m0 + (tid >> 1)) * wstride + k0 + BK], 0, 1);
            }

            // ---- Stage B (im2col): 128 pixels x 64 K, two k's per store.
#pragma unroll
            for (int e = tid; e < BN * BK / 2; e += 256) {
                const int n   = e >> 5;
                const int kk  = (e & 31) << 1;
                const int k   = k0 + kk;
                const int pix = n0 + n;
                const int b   = pix / HW;
                const int ph  = pix - b * HW;
                const int oh  = ph / H;
                const int ow  = ph - oh * H;
                float v0, v1;
                if (R3) {
                    const int c0 = k / 9;
                    const int r0 = k - c0 * 9;
                    int c1 = c0, r1 = r0 + 1;
                    if (r1 == 9) { r1 = 0; ++c1; }
                    const int ih0 = oh + r0 / 3 - 1, iw0 = ow + r0 % 3 - 1;
                    const int ih1 = oh + r1 / 3 - 1, iw1 = ow + r1 % 3 - 1;
                    v0 = ((unsigned)ih0 < (unsigned)H && (unsigned)iw0 < (unsigned)H)
                             ? src[(((size_t)b * C + c0) * H + ih0) * H + iw0] : 0.0f;
                    v1 = ((unsigned)ih1 < (unsigned)H && (unsigned)iw1 < (unsigned)H)
                             ? src[(((size_t)b * C + c1) * H + ih1) * H + iw1] : 0.0f;
                } else {
                    const float* sp = &src[(((size_t)b * C + k) * H + oh) * H + ow];
                    v0 = sp[0];
                    v1 = sp[HW];
                }
                *reinterpret_cast<unsigned*>(&Bs[n][kk]) = f2bf(v0) | (f2bf(v1) << 16);
            }

            if constexpr (WB16)
                asm volatile("s_wait_asynccnt 0x0" ::: "memory");
            __syncthreads();

            // ---- Fragments per ISA VGPR layouts; 2 K-steps x 2x4 tiles.
            // A 16x32 bf16: lanes 0-15 row M=l, K 0-7 then 16-23; lanes 16-31
            // same rows, K 8-15 then 24-31.
            // B 32x16 bf16: lanes 0-15 col N=l, K 0-7 then 16-23; lanes 16-31 K 8-15 then 24-31.
            const int kbA = (lane < 16) ? 0 : 8;
            const int kbB = (lane < 16) ? 0 : 8;
#pragma unroll
            for (int ks = 0; ks < BK; ks += 32) {
                v16bf afrag[2], bfrag[4];
#pragma unroll
                for (int t = 0; t < 2; ++t) {
                    const int row = wm * 32 + t * 16 + (lane & 15);
                    U8x32 ta;
                    ta.lo = *reinterpret_cast<const uint4v*>(&As[row][ks + kbA]);
                    ta.hi = *reinterpret_cast<const uint4v*>(&As[row][ks + kbA + 16]);
                    afrag[t] = __builtin_bit_cast(v16bf, ta);
                }
#pragma unroll
                for (int t = 0; t < 4; ++t) {
                    const int col = wn * 64 + t * 16 + (lane & 15);
                    U8x32 tb;
                    tb.lo = *reinterpret_cast<const uint4v*>(&Bs[col][ks + kbB]);
                    tb.hi = *reinterpret_cast<const uint4v*>(&Bs[col][ks + kbB + 16]);
                    bfrag[t] = __builtin_bit_cast(v16bf, tb);
                }
#pragma unroll
                for (int tm = 0; tm < 2; ++tm)
#pragma unroll
                    for (int tn = 0; tn < 4; ++tn)
                        acc[tm][tn] = __builtin_amdgcn_wmma_f32_16x16x32_bf16(
                            false, afrag[tm], false, bfrag[tn],
                            (short)0, acc[tm][tn], false, false);
            }
        }
    }

    // ---- Epilogue: C/D layout -> VGPR i holds M = i (lanes 0-15) / 8+i
    // (lanes 16-31), N = lane&15. Add bias, scatter to NCHW.
#pragma unroll
    for (int tm = 0; tm < 2; ++tm) {
#pragma unroll
        for (int tn = 0; tn < 4; ++tn) {
            const int col = n0 + wn * 64 + tn * 16 + (lane & 15);
            const int b   = col / HW;
            const int ph  = col - b * HW;
#pragma unroll
            for (int i = 0; i < 8; ++i) {
                const int ch = m0 + wm * 32 + tm * 16 + i + ((lane < 16) ? 0 : 8);
                const float vb = (biasA ? biasA[ch] : 0.0f) + (biasB ? biasB[ch] : 0.0f);
                out[((size_t)b * M + ch) * HW + ph] = acc[tm][tn][i] + vb;
            }
        }
    }
}

// ---------------------------------------------------------------------------
// Direct 3x3 conv 256 -> 16 (pad 1). Tiny vs. main GEMMs; weights fit in L2.
// ---------------------------------------------------------------------------
template<int H>
__global__ __launch_bounds__(256) void conv3x3_256_16(
    const float* __restrict__ u, const float* __restrict__ w,
    const float* __restrict__ bias, float* __restrict__ out)
{
    constexpr int HW = H * H;
    const int idx = blockIdx.x * 256 + threadIdx.x;      // over B*16*HW
    if (idx >= 2 * 16 * HW) return;
    const int ow = idx % H;
    int t = idx / H;
    const int oh = t % H; t /= H;
    const int oc = t % 16;
    const int b  = t / 16;

    float acc = bias[oc];
    for (int c = 0; c < 256; ++c) {
        const float* up = u + ((size_t)b * 256 + c) * HW;
        const float* wp = w + ((size_t)oc * 256 + c) * 9;
#pragma unroll
        for (int r = 0; r < 9; ++r) {
            const int ih = oh + r / 3 - 1;
            const int iw = ow + r % 3 - 1;
            if ((unsigned)ih < (unsigned)H && (unsigned)iw < (unsigned)H)
                acc += wp[r] * up[ih * H + iw];
        }
    }
    out[idx] = acc;
}

// ---------------------------------------------------------------------------
// 1x1 conv 16->2 + sigmoid + normalize: per-pixel blend weights.
// ---------------------------------------------------------------------------
template<int H>
__global__ __launch_bounds__(256) void u3_kernel(
    const float* __restrict__ u2, const float* __restrict__ w3,
    const float* __restrict__ b3, float* __restrict__ uw)
{
    constexpr int HW = H * H;
    const int pix = blockIdx.x * 256 + threadIdx.x;      // over B*HW
    if (pix >= 2 * HW) return;
    const int b  = pix / HW;
    const int ph = pix - b * HW;
    float s0 = b3[0], s1 = b3[1];
#pragma unroll
    for (int j = 0; j < 16; ++j) {
        const float v = u2[((size_t)b * 16 + j) * HW + ph];
        s0 += w3[j] * v;
        s1 += w3[16 + j] * v;
    }
    s0 = sigmoidf(s0);
    s1 = sigmoidf(s1);
    const float inv = 1.0f / (s0 + s1);
    uw[(size_t)pix * 2 + 0] = s0 * inv;
    uw[(size_t)pix * 2 + 1] = s1 * inv;
}

// ---------------------------------------------------------------------------
// update_feat = new_feat * w0 + feat2 * w1
// ---------------------------------------------------------------------------
template<int C, int H>
__global__ __launch_bounds__(256) void blend_kernel(
    const float* __restrict__ new_feat, const float* __restrict__ feat2,
    const float* __restrict__ uw, float* __restrict__ nf_out)
{
    constexpr int HW = H * H;
    const size_t idx = (size_t)blockIdx.x * 256 + threadIdx.x;  // over B*C*HW
    const int ph = (int)(idx % HW);
    const size_t t = idx / HW;
    const int b = (int)(t / C);
    const int pix = b * HW + ph;
    const float w0 = uw[(size_t)pix * 2 + 0];
    const float w1 = uw[(size_t)pix * 2 + 1];
    nf_out[idx] = new_feat[idx] * w0 + feat2[idx] * w1;
}

// ---------------------------------------------------------------------------
// LSTM pointwise gates: cx (B,4C,H,W) + c_pre -> h_next, c_next
// ---------------------------------------------------------------------------
template<int C, int H>
__global__ __launch_bounds__(256) void lstm_kernel(
    const float* __restrict__ cx, const float* __restrict__ c_pre,
    float* __restrict__ h_out, float* __restrict__ c_out)
{
    constexpr int HW = H * H;
    const size_t idx = (size_t)blockIdx.x * 256 + threadIdx.x;  // over B*C*HW
    const int ph = (int)(idx % HW);
    const size_t t = idx / HW;
    const int c = (int)(t % C);
    const int b = (int)(t / C);
    const size_t base = (size_t)b * 4 * C * HW + ph;
    const float i_ = cx[base + (size_t)c * HW];
    const float f_ = cx[base + (size_t)(C + c) * HW];
    const float o_ = cx[base + (size_t)(2 * C + c) * HW];
    const float g_ = cx[base + (size_t)(3 * C + c) * HW];
    const float cn = sigmoidf(f_) * c_pre[idx] + sigmoidf(i_) * tanhf(g_);
    h_out[idx] = sigmoidf(o_) * tanhf(cn);
    c_out[idx] = cn;
}

// ---------------------------------------------------------------------------
// Per-level driver
// ---------------------------------------------------------------------------
template<int C, int H>
static void run_level(void* const* din, float* outh, float* outc, float* outnf,
                      float* nf_t, float* u_t, float* u2_t, float* uw_t, float* cx_t,
                      unsigned short* wbf_base, bool use_bf16, hipStream_t s)
{
    const float* h    = (const float*)din[0];
    const float* cpre = (const float*)din[1];
    const float* xp   = (const float*)din[2];
    const float* x    = (const float*)din[3];
    const float* u1w  = (const float*)din[4];
    const float* u1b  = (const float*)din[5];
    const float* u2w  = (const float*)din[6];
    const float* u2b  = (const float*)din[7];
    const float* u3w  = (const float*)din[8];
    const float* u3b  = (const float*)din[9];
    const float* xw   = (const float*)din[10];
    const float* xb   = (const float*)din[11];
    const float* hw   = (const float*)din[12];
    const float* hb   = (const float*)din[13];
    constexpr int HW = H * H;
    constexpr int NP = 2 * HW;          // B*HW
    constexpr int WX = 4 * C * 9 * C;   // xw / hw element count
    constexpr int WU = 256 * 2 * C;     // u1w element count

    attn_kernel<C, H><<<NP / 8, 256, 0, s>>>(xp, x, nf_t);

    if (use_bf16) {
        unsigned short* wbf_x  = wbf_base;
        unsigned short* wbf_h  = wbf_x + (size_t)WX;
        unsigned short* wbf_u1 = wbf_h + (size_t)WX;
        f32_to_bf16_kernel<<<(WX / 4 + 255) / 256, 256, 0, s>>>(xw, wbf_x, WX / 4);
        f32_to_bf16_kernel<<<(WX / 4 + 255) / 256, 256, 0, s>>>(hw, wbf_h, WX / 4);
        f32_to_bf16_kernel<<<(WU / 4 + 255) / 256, 256, 0, s>>>(u1w, wbf_u1, WU / 4);

        wmma_gemm_kernel<C, H, 0, true><<<dim3(2, NP / 128), 256, 0, s>>>(
            nf_t, x, wbf_u1, wbf_u1 + C, 2 * C, u1b, nullptr, 256, u_t);

        conv3x3_256_16<H><<<(2 * 16 * HW) / 256, 256, 0, s>>>(u_t, u2w, u2b, u2_t);
        u3_kernel<H><<<(NP + 255) / 256, 256, 0, s>>>(u2_t, u3w, u3b, uw_t);
        blend_kernel<C, H><<<(2 * C * HW) / 256, 256, 0, s>>>(nf_t, x, uw_t, outnf);

        wmma_gemm_kernel<C, H, 1, true><<<dim3(4 * C / 128, NP / 128), 256, 0, s>>>(
            outnf, h, wbf_x, wbf_h, 9 * C, xb, hb, 4 * C, cx_t);
    } else {
        wmma_gemm_kernel<C, H, 0, false><<<dim3(2, NP / 128), 256, 0, s>>>(
            nf_t, x, u1w, u1w + C, 2 * C, u1b, nullptr, 256, u_t);

        conv3x3_256_16<H><<<(2 * 16 * HW) / 256, 256, 0, s>>>(u_t, u2w, u2b, u2_t);
        u3_kernel<H><<<(NP + 255) / 256, 256, 0, s>>>(u2_t, u3w, u3b, uw_t);
        blend_kernel<C, H><<<(2 * C * HW) / 256, 256, 0, s>>>(nf_t, x, uw_t, outnf);

        wmma_gemm_kernel<C, H, 1, false><<<dim3(4 * C / 128, NP / 128), 256, 0, s>>>(
            outnf, h, xw, hw, 9 * C, xb, hb, 4 * C, cx_t);
    }

    lstm_kernel<C, H><<<(2 * C * HW) / 256, 256, 0, s>>>(cx_t, cpre, outh, outc);
}

extern "C" void kernel_launch(void* const* d_in, const int* /*in_sizes*/, int /*n_in*/,
                              void* d_out, int /*out_size*/, void* d_ws, size_t ws_size,
                              hipStream_t stream)
{
    float* out = (float*)d_out;
    float* ws  = (float*)d_ws;

    const size_t S0 = (size_t)2 * 512 * 64 * 64;    // 4,194,304
    const size_t S1 = (size_t)2 * 1024 * 32 * 32;   // 2,097,152

    // Workspace layout (floats), reused sequentially by the two levels:
    float* nf_t = ws;                      // max B*C*HW           = 4,194,304
    float* u_t  = nf_t + 4194304;          // B*256*HW (level0)    = 2,097,152
    float* u2_t = u_t + 2097152;           // B*16*HW (level0)     =   131,072
    float* uw_t = u2_t + 131072;           // B*HW*2 (level0)      =    16,384
    float* cx_t = uw_t + 16384;            // B*4C*HW (level0)     = 16,777,216
    const size_t base_floats = 23216128;   // = sum of the above

    // bf16 weight staging region (reused per level); level-1 is the max:
    //   2 * (4096*9216) + 256*2048 = 76,021,760 ushorts = 152,043,520 bytes
    unsigned short* wbf = (unsigned short*)(ws + base_floats);
    const size_t need = base_floats * 4 + 152043520;
    const bool use_bf16 = (ws_size >= need);

    // Output layout: hn0 | cn0 | hn1 | cn1 | nf0 | nf1
    run_level<512, 64>(d_in,       out + 0,      out + S0,          out + 2 * S0 + 2 * S1,
                       nf_t, u_t, u2_t, uw_t, cx_t, wbf, use_bf16, stream);
    run_level<1024, 32>(d_in + 14, out + 2 * S0, out + 2 * S0 + S1, out + 3 * S0 + 2 * S1,
                       nf_t, u_t, u2_t, uw_t, cx_t, wbf, use_bf16, stream);
}
